// ViTMoE_8581344658169
// MI455X (gfx1250) — compile-verified, matrix operand order checked by reading, not run on hardware
//
#include <hip/hip_runtime.h>
#include <hip/hip_bf16.h>
#include <math.h>

// ---------------------------------------------------------------------------
// ViT-MoE forward, CDNA5 (gfx1250, wave32) WMMA implementation.
// GEMMs: f16 in, f32 acc via v_wmma_f32_16x16x32_f16; B tiles staged through
// LDS with the Tensor Data Mover (tensor_load_to_lds, TENSORcnt), double
// buffered so the DMA of k-step i+1 overlaps the WMMAs of step i.
// ---------------------------------------------------------------------------

typedef __attribute__((ext_vector_type(16))) _Float16     v16h;
typedef __attribute__((ext_vector_type(8)))  float        v8f;
typedef __attribute__((ext_vector_type(2)))  _Float16     h2v;
typedef __attribute__((ext_vector_type(4)))  unsigned int u32x4;
typedef __attribute__((ext_vector_type(8)))  int          i32x8;
typedef __attribute__((ext_vector_type(4)))  int          i32x4;

#define DEV __device__ __forceinline__

namespace {
constexpr int BB     = 128;
constexpr int EMB    = 384;
constexpr int HEADS  = 6;
constexpr int HD     = 64;
constexpr int NPATCH = 196;
constexpr int NTOK   = 197;
constexpr int HID    = 1536;
constexpr int NE     = 4;
constexpr int NCLS   = 1000;
constexpr int MTOK   = BB * NTOK;    // 25216 = 1576*16, /128 = 197
constexpr int MPATCH = BB * NPATCH;  // 25088 = 1568*16, /128 = 196
constexpr int KPATCH = 768;          // 3*16*16
constexpr int NTOKP  = 224;          // padded token count for attention (14*16)
constexpr int QKVN   = 3 * EMB;      // 1152
}

// ---- WMMA fragment helpers -------------------------------------------------
// A-fragment (16x32 f16, row-major source, ld = row stride):
//   lane = h*16 + m  (m = row, h = half)
//   vgpr v (0..7): grp=v>>2, vv=v&3 -> k = grp*16 + h*8 + vv*2 (pair)
// B-fragment of B[k][n] is loaded as the A-fragment of B^T, i.e. from a
// [N][K] row-major array with "row" = n.
DEV v16h load_frag(const _Float16* p, int ld, int k0, int lane) {
  const int m    = lane & 15;
  const int half = (lane >> 4) & 1;
  const _Float16* r = p + (size_t)m * ld + k0 + half * 8;
  v16h a;
#pragma unroll
  for (int v = 0; v < 8; ++v) {
    const int grp = v >> 2, vv = v & 3;
    h2v t = *(const h2v*)(r + grp * 16 + vv * 2);
    a[2 * v]     = t.x;
    a[2 * v + 1] = t.y;
  }
  return a;
}

DEV v8f wmma_f16(v16h a, v16h b, v8f c) {
  return __builtin_amdgcn_wmma_f32_16x16x32_f16(false, a, false, b, (short)0,
                                                c, false, false);
}

// ---- TDM: DMA one 64(rows) x 32(k) f16 tile, global -> LDS -----------------
// D# per CDNA5 ISA ch.8: group0 = {count, lds_addr, global_addr, type=2},
// group1 = {data_size=2B, tensor_dim0/1, tile_dim0=32, tile_dim1=64,
//           tensor_dim0_stride = ldw}. 2D tile: groups 2/3 zero.
// This toolchain exposes the 6-arg builtin (extra int32x8 before cpol).
DEV void tdm_load_b64x32(const _Float16* gsrc, _Float16* ldst, int ldw,
                         int rows_rem) {
  unsigned long long ga = (unsigned long long)(uintptr_t)gsrc;
  unsigned lds_off = (unsigned)(uintptr_t)ldst;  // generic LDS ptr: low 32 bits
  u32x4 g0;
  g0[0] = 1u;                                    // count=1, gather off
  g0[1] = lds_off;                               // lds_addr (bytes)
  g0[2] = (unsigned)ga;                          // global_addr[31:0]
  g0[3] = (unsigned)((ga >> 32) & 0x01ffffffu) | 0x80000000u;  // addr hi, type=2
  const unsigned td0 = (unsigned)ldw;            // tensor_dim0 (k elements)
  const unsigned td1 = (unsigned)rows_rem;       // tensor_dim1 (rows)
  const unsigned long long st0 = (unsigned)ldw;  // dim0 stride (elements)
  i32x8 g1;
  g1[0] = 0x00010000;                                            // data_size=1 (2B)
  g1[1] = (int)((td0 & 0xffffu) << 16);                          // dim0[15:0]
  g1[2] = (int)(((td0 >> 16) & 0xffffu) | ((td1 & 0xffffu) << 16));
  g1[3] = (int)(((td1 >> 16) & 0xffffu) | (32u << 16));          // tile_dim0=32
  g1[4] = 64;                                                    // tile_dim1=64
  g1[5] = (int)(unsigned)(st0 & 0xffffffffu);
  g1[6] = (int)((st0 >> 32) & 0xffffu);
  g1[7] = 0;
  i32x4 gz4 = {0, 0, 0, 0};
  i32x8 gz8 = {0, 0, 0, 0, 0, 0, 0, 0};
  __builtin_amdgcn_tensor_load_to_lds(g0, g1, gz4, gz4, gz8, 0);
}

// One wave computes a 16(M) x 64(N) tile; the block's 8 waves share the 64-col
// B tile, DMA'd by wave 0 via TDM into double-buffered LDS.
// A: [M][K] f16 row-major (direct global b128 loads).
// W: [N][K] f16 row-major, wrows = total rows of W (>= col0+64).
DEV void gemm16x64_tdm(const _Float16* A, int lda, const _Float16* W, int ldw,
                       int wrows, int row0, int col0, int K,
                       _Float16 (*Bsm)[64 * 32], v8f acc[4]) {
  const int lane = threadIdx.x & 31;
  const int wave = threadIdx.x >> 5;
  const int nk   = K >> 5;
  const _Float16* wbase = W + (size_t)col0 * ldw;
  if (wave == 0) tdm_load_b64x32(wbase, Bsm[0], ldw, wrows - col0);
  for (int i = 0; i < nk; ++i) {
    if (wave == 0) __builtin_amdgcn_s_wait_tensorcnt(0);
    __syncthreads();  // publish Bsm[i&1]
    if (wave == 0 && i + 1 < nk)
      tdm_load_b64x32(wbase + (i + 1) * 32, Bsm[(i + 1) & 1], ldw,
                      wrows - col0);
    v16h a = load_frag(A + (size_t)row0 * lda, lda, i * 32, lane);
    const _Float16* bs = Bsm[i & 1];
#pragma unroll
    for (int t = 0; t < 4; ++t) {
      v16h b = load_frag(bs + t * 16 * 32, 32, 0, lane);
      acc[t] = wmma_f16(a, b, acc[t]);
    }
    __syncthreads();  // done reading Bsm[i&1] before it is re-filled
  }
}

// ---- small utility kernels -------------------------------------------------
__global__ void k_cast_half(const float* s, _Float16* d, int n) {
  for (int i = blockIdx.x * blockDim.x + threadIdx.x; i < n;
       i += gridDim.x * blockDim.x)
    d[i] = (_Float16)s[i];
}

__global__ void k_zero_half(_Float16* d, int n) {
  for (int i = blockIdx.x * blockDim.x + threadIdx.x; i < n;
       i += gridDim.x * blockDim.x)
    d[i] = (_Float16)0.0f;
}

// s[e][k][n] (f32) -> d[e][n][k] (f16)
__global__ void k_transpose_half(const float* s, _Float16* d, int Kd, int Nd,
                                 int ne) {
  int total = ne * Kd * Nd;
  for (int i = blockIdx.x * blockDim.x + threadIdx.x; i < total;
       i += gridDim.x * blockDim.x) {
    int e   = i / (Kd * Nd);
    int rem = i - e * (Kd * Nd);
    int k   = rem / Nd;
    int n   = rem - k * Nd;
    d[((size_t)e * Nd + n) * Kd + k] = (_Float16)s[i];
  }
}

// im2col for patch embedding: A[b*196+p][c*256+i*16+j] = x[b,c,py*16+i,px*16+j]
__global__ void k_im2col(const float* x, _Float16* a) {
  int total = MPATCH * KPATCH;
  for (int i = blockIdx.x * blockDim.x + threadIdx.x; i < total;
       i += gridDim.x * blockDim.x) {
    int row = i / KPATCH;
    int k   = i - row * KPATCH;
    int b   = row / NPATCH;
    int p   = row - b * NPATCH;
    int py  = p / 14, px = p - py * 14;
    int c   = k >> 8;
    int rr  = (k >> 4) & 15;
    int cc  = k & 15;
    size_t src = (((size_t)b * 3 + c) * 224 + (py * 16 + rr)) * 224 +
                 (px * 16 + cc);
    a[i] = (_Float16)x[src];
  }
}

// ---- patch-embed GEMM: h[b*197+1+p][n] = A @ conv_w^T + conv_b -------------
__global__ void k_gemm_patch(const _Float16* A, const _Float16* W,
                             const float* bias, float* h) {
  __shared__ _Float16 Bsm[2][64 * 32];
  const int lane = threadIdx.x & 31, wave = threadIdx.x >> 5;
  const int row0 = (blockIdx.y * 8 + wave) * 16;
  const int col0 = blockIdx.x * 64;
  v8f acc[4] = {};
  gemm16x64_tdm(A, KPATCH, W, KPATCH, EMB, row0, col0, KPATCH, Bsm, acc);
  const int cn = lane & 15, half = lane >> 4;
#pragma unroll
  for (int t = 0; t < 4; ++t)
#pragma unroll
    for (int r = 0; r < 8; ++r) {
      int row = row0 + r + 8 * half;
      int col = col0 + t * 16 + cn;
      int b   = row / NPATCH, p = row - b * NPATCH;
      size_t tok = (size_t)b * NTOK + 1 + p;
      h[tok * EMB + col] = acc[t][r] + bias[col];
    }
}

// cls token rows + pos embedding
__global__ void k_poscls(float* h, const float* cls, const float* pos) {
  int total = MTOK * EMB;
  for (int i = blockIdx.x * blockDim.x + threadIdx.x; i < total;
       i += gridDim.x * blockDim.x) {
    int t = i / EMB, e = i - t * EMB;
    int tok = t % NTOK;
    if (tok == 0)
      h[i] = cls[e] + pos[e];
    else
      h[i] += pos[(size_t)tok * EMB + e];
  }
}

// ---- LayerNorm: wave per row, 384 = 12 elems/lane --------------------------
__global__ void k_ln(const float* in, const float* g, const float* bb,
                     _Float16* out16, float* out32, int M) {
  const int lane = threadIdx.x & 31, wave = threadIdx.x >> 5;
  const int row  = blockIdx.x * 8 + wave;
  if (row >= M) return;
  const float* p = in + (size_t)row * EMB;
  float vals[12];
  float s = 0.0f;
#pragma unroll
  for (int i = 0; i < 12; ++i) {
    vals[i] = p[lane + i * 32];
    s += vals[i];
  }
#pragma unroll
  for (int m = 16; m >= 1; m >>= 1) s += __shfl_xor(s, m, 32);
  float mean = s * (1.0f / EMB);
  float v = 0.0f;
#pragma unroll
  for (int i = 0; i < 12; ++i) {
    float d = vals[i] - mean;
    v += d * d;
  }
#pragma unroll
  for (int m = 16; m >= 1; m >>= 1) v += __shfl_xor(v, m, 32);
  float rs = rsqrtf(v * (1.0f / EMB) + 1e-5f);
#pragma unroll
  for (int i = 0; i < 12; ++i) {
    int c = lane + i * 32;
    float o = (vals[i] - mean) * rs * g[c] + bb[c];
    out16[(size_t)row * EMB + c] = (_Float16)o;
    if (out32) out32[(size_t)row * EMB + c] = o;
  }
}

// LN2 applied only to the 128 cls rows -> cls16[128][384]
__global__ void k_ln_cls(const float* in, const float* g, const float* bb,
                         _Float16* out16) {
  const int lane = threadIdx.x & 31, wave = threadIdx.x >> 5;
  const int b    = blockIdx.x * 8 + wave;
  if (b >= BB) return;
  const float* p = in + (size_t)b * NTOK * EMB;  // token 0 of batch b
  float vals[12];
  float s = 0.0f;
#pragma unroll
  for (int i = 0; i < 12; ++i) {
    vals[i] = p[lane + i * 32];
    s += vals[i];
  }
#pragma unroll
  for (int m = 16; m >= 1; m >>= 1) s += __shfl_xor(s, m, 32);
  float mean = s * (1.0f / EMB);
  float v = 0.0f;
#pragma unroll
  for (int i = 0; i < 12; ++i) {
    float d = vals[i] - mean;
    v += d * d;
  }
#pragma unroll
  for (int m = 16; m >= 1; m >>= 1) v += __shfl_xor(v, m, 32);
  float rs = rsqrtf(v * (1.0f / EMB) + 1e-5f);
#pragma unroll
  for (int i = 0; i < 12; ++i) {
    int c = lane + i * 32;
    out16[(size_t)b * EMB + c] = (_Float16)((vals[i] - mean) * rs * g[c] + bb[c]);
  }
}

// ---- QKV GEMM with scatter epilogue ----------------------------------------
// Q,K: [B,H,NTOKP,64] f16   V^T: [B,H,64,NTOKP] f16  (pad rows pre-zeroed)
__global__ void k_gemm_qkv(const _Float16* A, const _Float16* W,
                           const float* bias, _Float16* Qb, _Float16* Kb,
                           _Float16* VT) {
  __shared__ _Float16 Bsm[2][64 * 32];
  const int lane = threadIdx.x & 31, wave = threadIdx.x >> 5;
  const int row0 = (blockIdx.y * 8 + wave) * 16;
  const int col0 = blockIdx.x * 64;
  v8f acc[4] = {};
  gemm16x64_tdm(A, EMB, W, EMB, QKVN, row0, col0, EMB, Bsm, acc);
  const int cn = lane & 15, half = lane >> 4;
#pragma unroll
  for (int t = 0; t < 4; ++t)
#pragma unroll
    for (int r = 0; r < 8; ++r) {
      int row = row0 + r + 8 * half;            // flat token
      int col = col0 + t * 16 + cn;             // 0..1151
      float v = acc[t][r] + bias[col];
      int b   = row / NTOK, tok = row - b * NTOK;
      int sec = col / EMB, w = col - sec * EMB;
      int h   = w / HD, d = w - h * HD;
      size_t bh = (size_t)b * HEADS + h;
      if (sec == 0)
        Qb[(bh * NTOKP + tok) * HD + d] = (_Float16)v;
      else if (sec == 1)
        Kb[(bh * NTOKP + tok) * HD + d] = (_Float16)v;
      else
        VT[(bh * HD + d) * NTOKP + tok] = (_Float16)v;
    }
}

// ---- Flash-style attention: block per (b,h), wave per 16-row q-tile --------
__global__ void k_attention(const _Float16* Qb, const _Float16* Kb,
                            const _Float16* VT, _Float16* attn) {
  __shared__ _Float16 Plds[8][16 * 32];
  const int lane = threadIdx.x & 31, wave = threadIdx.x >> 5;
  const int bh   = blockIdx.x;
  const int b    = bh / HEADS, h = bh - b * HEADS;
  const _Float16* Q = Qb + (size_t)bh * NTOKP * HD;
  const _Float16* K = Kb + (size_t)bh * NTOKP * HD;
  const _Float16* V = VT + (size_t)bh * HD * NTOKP;
  const int cn = lane & 15, half = lane >> 4;
  _Float16* pl = &Plds[wave][0];

  for (int qt = wave; qt < NTOKP / 16; qt += 8) {
    const int q0 = qt * 16;
    v16h qf0 = load_frag(Q + (size_t)q0 * HD, HD, 0, lane);
    v16h qf1 = load_frag(Q + (size_t)q0 * HD, HD, 32, lane);
    v8f o[4] = {};
    float mrow[8], lrow[8];
#pragma unroll
    for (int r = 0; r < 8; ++r) {
      mrow[r] = -1e30f;
      lrow[r] = 0.0f;
    }
    for (int kc = 0; kc < NTOKP; kc += 32) {
      // two 16x16 S tiles (columns kc..kc+15, kc+16..kc+31)
      v8f s0 = {}, s1 = {};
      {
        v16h b00 = load_frag(K + (size_t)kc * HD, HD, 0, lane);
        s0 = wmma_f16(qf0, b00, s0);
        v16h b01 = load_frag(K + (size_t)kc * HD, HD, 32, lane);
        s0 = wmma_f16(qf1, b01, s0);
        v16h b10 = load_frag(K + (size_t)(kc + 16) * HD, HD, 0, lane);
        s1 = wmma_f16(qf0, b10, s1);
        v16h b11 = load_frag(K + (size_t)(kc + 16) * HD, HD, 32, lane);
        s1 = wmma_f16(qf1, b11, s1);
      }
      // scale + key-pad mask
      const float sc = 0.125f;  // 1/sqrt(64)
#pragma unroll
      for (int r = 0; r < 8; ++r) {
        float a0 = s0[r] * sc, a1 = s1[r] * sc;
        if (kc + cn >= NTOK) a0 = -1e30f;
        if (kc + 16 + cn >= NTOK) a1 = -1e30f;
        s0[r] = a0;
        s1[r] = a1;
      }
      // online softmax update (row = r + 8*half, spread over 16 lanes)
#pragma unroll
      for (int r = 0; r < 8; ++r) {
        float m = fmaxf(s0[r], s1[r]);
#pragma unroll
        for (int msk = 8; msk >= 1; msk >>= 1) m = fmaxf(m, __shfl_xor(m, msk, 32));
        float newm  = fmaxf(mrow[r], m);
        float scale = __expf(mrow[r] - newm);
        mrow[r]     = newm;
        lrow[r] *= scale;
#pragma unroll
        for (int t = 0; t < 4; ++t) o[t][r] *= scale;
        float p0 = __expf(s0[r] - newm);
        float p1 = __expf(s1[r] - newm);
        s0[r] = p0;
        s1[r] = p1;
        float ps = p0 + p1;
#pragma unroll
        for (int msk = 8; msk >= 1; msk >>= 1) ps += __shfl_xor(ps, msk, 32);
        lrow[r] += ps;
      }
      // P -> LDS [16 rows][32 cols] then re-load as A-fragment
#pragma unroll
      for (int r = 0; r < 8; ++r) {
        pl[(r + 8 * half) * 32 + cn]      = (_Float16)s0[r];
        pl[(r + 8 * half) * 32 + 16 + cn] = (_Float16)s1[r];
      }
      asm volatile("s_wait_dscnt 0x0" ::: "memory");
      v16h pf = load_frag(pl, 32, 0, lane);
#pragma unroll
      for (int t = 0; t < 4; ++t) {
        v16h vf = load_frag(V + (size_t)(t * 16) * NTOKP, NTOKP, kc, lane);
        o[t] = wmma_f16(pf, vf, o[t]);
      }
    }
    // store O / l  -> attn[b*197+q][h*64 + d]
#pragma unroll
    for (int t = 0; t < 4; ++t)
#pragma unroll
      for (int r = 0; r < 8; ++r) {
        int q = q0 + r + 8 * half;
        if (q < NTOK)
          attn[((size_t)(b * NTOK + q)) * EMB + h * HD + t * 16 + cn] =
              (_Float16)(o[t][r] / lrow[r]);
      }
  }
}

// ---- generic GEMM + bias -> f32 (out-proj, head) ---------------------------
__global__ void k_gemm_bias_f32(const _Float16* A, int lda, const _Float16* W,
                                int ldw, int wrows, const float* bias, float* C,
                                int ldc, int K, int Nmax) {
  __shared__ _Float16 Bsm[2][64 * 32];
  const int lane = threadIdx.x & 31, wave = threadIdx.x >> 5;
  const int row0 = (blockIdx.y * 8 + wave) * 16;
  const int col0 = blockIdx.x * 64;
  v8f acc[4] = {};
  gemm16x64_tdm(A, lda, W, ldw, wrows, row0, col0, K, Bsm, acc);
  const int cn = lane & 15, half = lane >> 4;
#pragma unroll
  for (int t = 0; t < 4; ++t)
#pragma unroll
    for (int r = 0; r < 8; ++r) {
      int col = col0 + t * 16 + cn;
      if (col < Nmax)
        C[(size_t)(row0 + r + 8 * half) * ldc + col] = acc[t][r] + bias[col];
    }
}

// ---- FFN1: GEMM + bias + exact GELU -> f16 ---------------------------------
__global__ void k_gemm_gelu_f16(const _Float16* A, const _Float16* W,
                                const float* bias, _Float16* C) {
  __shared__ _Float16 Bsm[2][64 * 32];
  const int lane = threadIdx.x & 31, wave = threadIdx.x >> 5;
  const int row0 = (blockIdx.y * 8 + wave) * 16;
  const int col0 = blockIdx.x * 64;
  v8f acc[4] = {};
  gemm16x64_tdm(A, EMB, W, EMB, HID, row0, col0, EMB, Bsm, acc);
  const int cn = lane & 15, half = lane >> 4;
#pragma unroll
  for (int t = 0; t < 4; ++t)
#pragma unroll
    for (int r = 0; r < 8; ++r) {
      int col = col0 + t * 16 + cn;
      float v = acc[t][r] + bias[col];
      v = 0.5f * v * (1.0f + erff(v * 0.70710678118654752f));
      C[(size_t)(row0 + r + 8 * half) * HID + col] = (_Float16)v;
    }
}

// ---- FFN2: GEMM + bias, weighted (0 / 0.5) accumulate into moe -------------
__global__ void k_gemm_moe_acc(const _Float16* A, const _Float16* W,
                               const float* bias, const float* w05, int e,
                               float* moe, int init) {
  __shared__ _Float16 Bsm[2][64 * 32];
  const int lane = threadIdx.x & 31, wave = threadIdx.x >> 5;
  const int row0 = (blockIdx.y * 8 + wave) * 16;
  const int col0 = blockIdx.x * 64;
  v8f acc[4] = {};
  gemm16x64_tdm(A, HID, W, HID, EMB, row0, col0, HID, Bsm, acc);
  const int cn = lane & 15, half = lane >> 4;
#pragma unroll
  for (int t = 0; t < 4; ++t)
#pragma unroll
    for (int r = 0; r < 8; ++r) {
      int row = row0 + r + 8 * half;
      int col = col0 + t * 16 + cn;
      float w = w05[(size_t)row * NE + e];
      float v = w * (acc[t][r] + bias[col]);
      size_t idx = (size_t)row * EMB + col;
      if (init)
        moe[idx] = v;
      else
        moe[idx] += v;
    }
}

// ---- router: logits -> top-2 -> 0.5/0 weights ------------------------------
__global__ void k_router(const float* h, const float* rw, const float* rb,
                         float* w05) {
  for (int t = blockIdx.x * blockDim.x + threadIdx.x; t < MTOK;
       t += gridDim.x * blockDim.x) {
    const float* hr = h + (size_t)t * EMB;
    float sc[NE];
#pragma unroll
    for (int e = 0; e < NE; ++e) {
      const float* wr = rw + (size_t)e * EMB;
      float s = rb[e];
      for (int k = 0; k < EMB; ++k) s += hr[k] * wr[k];
      sc[e] = s;  // softmax is monotonic: top-k on logits == top-k on softmax
    }
    int i0 = 0;
#pragma unroll
    for (int e = 1; e < NE; ++e)
      if (sc[e] > sc[i0]) i0 = e;
    int i1 = -1;
#pragma unroll
    for (int e = 0; e < NE; ++e)
      if (e != i0 && (i1 < 0 || sc[e] > sc[i1])) i1 = e;
#pragma unroll
    for (int e = 0; e < NE; ++e)
      w05[(size_t)t * NE + e] = (e == i0 || e == i1) ? 0.5f : 0.0f;
  }
}

// ---------------------------------------------------------------------------
extern "C" void kernel_launch(void* const* d_in, const int* in_sizes, int n_in,
                              void* d_out, int out_size, void* d_ws,
                              size_t ws_size, hipStream_t stream) {
  (void)in_sizes; (void)n_in; (void)out_size; (void)ws_size;
  const float* x          = (const float*)d_in[0];
  const float* conv_w     = (const float*)d_in[1];
  const float* conv_b     = (const float*)d_in[2];
  const float* cls_tok    = (const float*)d_in[3];
  const float* pos_emb    = (const float*)d_in[4];
  const float* ln1_g      = (const float*)d_in[5];
  const float* ln1_b      = (const float*)d_in[6];
  const float* in_proj_w  = (const float*)d_in[7];
  const float* in_proj_b  = (const float*)d_in[8];
  const float* out_proj_w = (const float*)d_in[9];
  const float* out_proj_b = (const float*)d_in[10];
  const float* router_w   = (const float*)d_in[11];
  const float* router_b   = (const float*)d_in[12];
  const float* ew1        = (const float*)d_in[13];
  const float* eb1        = (const float*)d_in[14];
  const float* ew2        = (const float*)d_in[15];
  const float* eb2        = (const float*)d_in[16];
  const float* ln2_g      = (const float*)d_in[17];
  const float* ln2_b      = (const float*)d_in[18];
  const float* head_w     = (const float*)d_in[19];
  const float* head_b     = (const float*)d_in[20];
  float* out = (float*)d_out;

  char* wsp = (char*)d_ws;
  auto alloc = [&](size_t bytes) -> char* {
    char* p = wsp;
    wsp += (bytes + 255) & ~(size_t)255;
    return p;
  };

  _Float16* cw16    = (_Float16*)alloc((size_t)EMB * KPATCH * 2);
  _Float16* wqkv16  = (_Float16*)alloc((size_t)QKVN * EMB * 2);
  _Float16* wout16  = (_Float16*)alloc((size_t)EMB * EMB * 2);
  _Float16* whead16 = (_Float16*)alloc((size_t)1024 * EMB * 2);  // padded N
  _Float16* ew1t    = (_Float16*)alloc((size_t)NE * HID * EMB * 2);
  _Float16* ew2t    = (_Float16*)alloc((size_t)NE * EMB * HID * 2);
  _Float16* apatch  = (_Float16*)alloc((size_t)MPATCH * KPATCH * 2);
  float*    h32     = (float*)   alloc((size_t)MTOK * EMB * 4);
  _Float16* hn16    = (_Float16*)alloc((size_t)MTOK * EMB * 2);
  _Float16* Qb      = (_Float16*)alloc((size_t)BB * HEADS * NTOKP * HD * 2);
  _Float16* Kb      = (_Float16*)alloc((size_t)BB * HEADS * NTOKP * HD * 2);
  _Float16* VT      = (_Float16*)alloc((size_t)BB * HEADS * HD * NTOKP * 2);
  _Float16* attn16  = (_Float16*)alloc((size_t)MTOK * EMB * 2);
  float*    o32     = (float*)   alloc((size_t)MTOK * EMB * 4);
  float*    h232    = (float*)   alloc((size_t)MTOK * EMB * 4);
  _Float16* h216    = (_Float16*)alloc((size_t)MTOK * EMB * 2);
  float*    w05     = (float*)   alloc((size_t)MTOK * NE * 4);
  _Float16* eh16    = (_Float16*)alloc((size_t)MTOK * HID * 2);
  float*    moe     = (float*)   alloc((size_t)MTOK * EMB * 4);
  _Float16* cls16   = (_Float16*)alloc((size_t)BB * EMB * 2);

  auto gs = [](int n) { return dim3((n + 255) / 256); };
  const dim3 T(256);

  // 1) weight conversions
  k_cast_half<<<gs(EMB * KPATCH), T, 0, stream>>>(conv_w, cw16, EMB * KPATCH);
  k_cast_half<<<gs(QKVN * EMB), T, 0, stream>>>(in_proj_w, wqkv16, QKVN * EMB);
  k_cast_half<<<gs(EMB * EMB), T, 0, stream>>>(out_proj_w, wout16, EMB * EMB);
  k_zero_half<<<gs(1024 * EMB), T, 0, stream>>>(whead16, 1024 * EMB);
  k_cast_half<<<gs(NCLS * EMB), T, 0, stream>>>(head_w, whead16, NCLS * EMB);
  k_transpose_half<<<gs(NE * EMB * HID), T, 0, stream>>>(ew1, ew1t, EMB, HID, NE);
  k_transpose_half<<<gs(NE * HID * EMB), T, 0, stream>>>(ew2, ew2t, HID, EMB, NE);

  // 2) patch embedding
  k_im2col<<<gs(MPATCH * KPATCH), T, 0, stream>>>(x, apatch);
  k_gemm_patch<<<dim3(EMB / 64, MPATCH / 128), T, 0, stream>>>(apatch, cw16,
                                                               conv_b, h32);
  k_poscls<<<gs(MTOK * EMB), T, 0, stream>>>(h32, cls_tok, pos_emb);

  // 3) LN1 -> f16
  k_ln<<<dim3((MTOK + 7) / 8), T, 0, stream>>>(h32, ln1_g, ln1_b, hn16,
                                               (float*)nullptr, MTOK);

  // 4) QKV (pad rows pre-zeroed so attention masking is clean)
  const int qkvcnt = BB * HEADS * NTOKP * HD;
  k_zero_half<<<gs(qkvcnt), T, 0, stream>>>(Qb, qkvcnt);
  k_zero_half<<<gs(qkvcnt), T, 0, stream>>>(Kb, qkvcnt);
  k_zero_half<<<gs(qkvcnt), T, 0, stream>>>(VT, qkvcnt);
  k_gemm_qkv<<<dim3(QKVN / 64, MTOK / 128), T, 0, stream>>>(hn16, wqkv16,
                                                            in_proj_b, Qb, Kb,
                                                            VT);

  // 5) attention
  k_attention<<<dim3(BB * HEADS), T, 0, stream>>>(Qb, Kb, VT, attn16);

  // 6) out-proj, second LN1 (same params, per reference)
  k_gemm_bias_f32<<<dim3(EMB / 64, MTOK / 128), T, 0, stream>>>(
      attn16, EMB, wout16, EMB, EMB, out_proj_b, o32, EMB, EMB, EMB);
  k_ln<<<dim3((MTOK + 7) / 8), T, 0, stream>>>(o32, ln1_g, ln1_b, h216, h232,
                                               MTOK);

  // 7) router top-2
  k_router<<<gs(MTOK), T, 0, stream>>>(h232, router_w, router_b, w05);

  // 8) dense 4-expert FFN with weighted accumulate (mean of top-2)
  for (int e = 0; e < NE; ++e) {
    k_gemm_gelu_f16<<<dim3(HID / 64, MTOK / 128), T, 0, stream>>>(
        h216, ew1t + (size_t)e * HID * EMB, eb1 + (size_t)e * HID, eh16);
    k_gemm_moe_acc<<<dim3(EMB / 64, MTOK / 128), T, 0, stream>>>(
        eh16, ew2t + (size_t)e * EMB * HID, eb2 + (size_t)e * EMB, w05, e, moe,
        e == 0 ? 1 : 0);
  }

  // 9) LN2 on cls rows + classification head
  k_ln_cls<<<dim3(BB / 8), T, 0, stream>>>(moe, ln2_g, ln2_b, cls16);
  k_gemm_bias_f32<<<dim3(16, 1), T, 0, stream>>>(cls16, EMB, whead16, EMB, 1024,
                                                 head_b, out, NCLS, EMB, NCLS);
}
